// REGNNCognitiveProcessor_47382079209915
// MI455X (gfx1250) — matrix operand order repaired and checked by reading, not compiled, not run on hardware
//
#include <hip/hip_runtime.h>
#include <hip/hip_bf16.h>

// ---------------------------------------------------------------------------
// REGNN cognitive processor pipeline for MI455X (gfx1250, wave32, WMMA bf16)
//   B=16, N_FRAMES=128, IN_DIM=256, OUT_NODES=512, N_CH=8, K_NEIGH=32
// d_out = node_features [16,512,128] f32  ||  edge [16,8,512,512] f32
// ---------------------------------------------------------------------------

typedef __attribute__((ext_vector_type(16))) __bf16 v16bf;
typedef __attribute__((ext_vector_type(8)))  __bf16 v8bf;
typedef __attribute__((ext_vector_type(2)))  __bf16 v2bf;
typedef __attribute__((ext_vector_type(8)))  float  v8f;
typedef __attribute__((ext_vector_type(4)))  float  v4f;

#define NB      16
#define NFRM    128
#define DIN     256
#define DOUT    512
#define NCH     8
#define KNEIGH  32

__device__ __forceinline__ v8f wmma_bf16(v16bf a, v16bf b, v8f c) {
  return __builtin_amdgcn_wmma_f32_16x16x32_bf16(false, a, false, b, (short)0, c,
                                                 false, false);
}

__device__ __forceinline__ v16bf cat16(v8bf lo, v8bf hi) {
  return __builtin_shufflevector(lo, hi, 0, 1, 2, 3, 4, 5, 6, 7, 8, 9, 10, 11,
                                 12, 13, 14, 15);
}

// A fragment: 16(M) x 32(K) row-major tile, leading dim ld (elements, 16B-aligned rows).
// lanes 0-15: row=lane, K {0..7,16..23}; lanes 16-31: row=lane-16, K {8..15,24..31}
__device__ __forceinline__ v16bf load_frag_a(const __bf16* p, int ld, int lane) {
  int row = lane & 15;
  int k0  = (lane >> 4) << 3;
  const __bf16* r = p + row * ld + k0;
  v8bf lo = *(const v8bf*)(r);
  v8bf hi = *(const v8bf*)(r + 16);
  return cat16(lo, hi);
}

// B fragment from B^T storage (N x K row-major, two 16B loads per lane, at k0 and k0+16).
// lanes 0-15: col=lane, K {0..7,16..23}; lanes 16-31: col=lane-16, K {8..15,24..31}
__device__ __forceinline__ v16bf load_frag_bt(const __bf16* p, int ld, int lane) {
  int col = lane & 15;
  int k0  = (lane >> 4) << 3;
  const __bf16* r = p + (size_t)col * ld + k0;
  v8bf lo = *(const v8bf*)(r);
  v8bf hi = *(const v8bf*)(r + 16);
  return cat16(lo, hi);
}

// Stage a 32 x C f32 slab (row-major, ld C) as bf16 TRANSPOSED into sWt[c*32+r].
// Each thread handles 2 rows x 8 cols: 4x float4 global loads + 8 packed b32 LDS
// stores (adjacent K pairs are adjacent in the transposed slab).
template <int C>
__device__ __forceinline__ void stage_wt_slab(const float* __restrict__ Wsrc,
                                              __bf16* sWt, int t) {
  const int groups = 16 * (C / 8);  // 2-row x 8-col micro-tiles
#pragma unroll
  for (int g0 = 0; g0 < groups; g0 += 256) {
    int g = g0 + t;
    int rp = g / (C / 8);
    int c8 = (g % (C / 8)) * 8;
    int r = rp * 2;
    const v4f* p0 = (const v4f*)(Wsrc + (size_t)r * C + c8);
    const v4f* p1 = (const v4f*)(Wsrc + (size_t)(r + 1) * C + c8);
    v4f a0 = p0[0], a1 = p0[1];
    v4f b0 = p1[0], b1 = p1[1];
    float r0[8] = {a0[0], a0[1], a0[2], a0[3], a1[0], a1[1], a1[2], a1[3]};
    float r1[8] = {b0[0], b0[1], b0[2], b0[3], b1[0], b1[1], b1[2], b1[3]};
#pragma unroll
    for (int j = 0; j < 8; ++j) {
      v2bf pk;
      pk[0] = (__bf16)r0[j];
      pk[1] = (__bf16)r1[j];
      *(v2bf*)(sWt + (size_t)(c8 + j) * 32 + r) = pk;
    }
  }
}

// ---------------------------------------------------------------------------
// K0: spectral norm (power iteration, 8 iters) for w1[n] (256x256) / w2[n] (256x512)
// ---------------------------------------------------------------------------
__global__ __launch_bounds__(256) void kn_specnorm(const float* __restrict__ w1,
                                                   const float* __restrict__ w2,
                                                   float* s1inv, float* s2inv) {
  int m = blockIdx.x;
  bool isW2 = (m >= NFRM);
  int n = isW2 ? m - NFRM : m;
  const float* W = isW2 ? (w2 + (size_t)n * DIN * DOUT)
                        : (w1 + (size_t)n * DIN * DIN);
  const int rows = DIN;
  const int cols = isW2 ? DOUT : DIN;
  __shared__ float su[DIN];
  __shared__ float sv[DOUT];
  __shared__ float red[256];
  int t = threadIdx.x;
  for (int j = t; j < cols; j += 256) sv[j] = 1.0f;
  __syncthreads();
  float sigma = 0.0f;
  for (int it = 0; it < 8; ++it) {
    {
      float acc = 0.0f;
      const float* r = W + (size_t)t * cols;
      for (int j = 0; j < cols; ++j) acc += r[j] * sv[j];
      su[t] = acc;
    }
    __syncthreads();
    red[t] = su[t] * su[t];
    __syncthreads();
    for (int s = 128; s > 0; s >>= 1) { if (t < s) red[t] += red[t + s]; __syncthreads(); }
    float un = sqrtf(red[0]) + 1e-20f;
    __syncthreads();
    su[t] = su[t] / un;
    __syncthreads();
    for (int j = t; j < cols; j += 256) {
      float acc = 0.0f;
      for (int i = 0; i < rows; ++i) acc += W[(size_t)i * cols + j] * su[i];
      sv[j] = acc;
    }
    __syncthreads();
    float l2 = 0.0f;
    for (int j = t; j < cols; j += 256) l2 += sv[j] * sv[j];
    red[t] = l2;
    __syncthreads();
    for (int s = 128; s > 0; s >>= 1) { if (t < s) red[t] += red[t + s]; __syncthreads(); }
    sigma = sqrtf(red[0]);
    __syncthreads();
    float inv = 1.0f / (sigma + 1e-20f);
    for (int j = t; j < cols; j += 256) sv[j] *= inv;
    __syncthreads();
  }
  if (t == 0) {
    float s = fmaxf(sigma, 1e-6f);
    (isW2 ? s2inv : s1inv)[n] = 1.0f / s;
  }
}

// ---------------------------------------------------------------------------
// wqk -> transposed bf16 copy: wqkT[c][k] = wqk[k][c]
// ---------------------------------------------------------------------------
__global__ void kn_cvt_wqkT(const float* __restrict__ src, __bf16* __restrict__ dstT) {
  int i = blockIdx.x * blockDim.x + threadIdx.x;  // over 2048*128
  int c = i >> 7, k = i & 127;
  dstT[i] = (__bf16)src[(size_t)k * 2048 + c];
}

// ---------------------------------------------------------------------------
// K1: per-frame MLP -> conv[b][frame][node] f32 (coalesced scratch).
// ---------------------------------------------------------------------------
__global__ __launch_bounds__(256) void kn_mlp(
    const float* __restrict__ x, const float* __restrict__ w1,
    const float* __restrict__ b1, const float* __restrict__ w2,
    const float* __restrict__ b2, const float* __restrict__ s1inv,
    const float* __restrict__ s2inv, float* __restrict__ conv) {
  const int n = blockIdx.x;  // frame
  __shared__ __bf16 sX[16 * DIN];    // 8 KB
  __shared__ __bf16 sWt[DOUT * 32];  // 32 KB, [col][k] transposed slab
  __shared__ __bf16 sH[16 * DIN];    // 8 KB
  int t = threadIdx.x;
  int lane = t & 31, wv = t >> 5;
  for (int i = t; i < 16 * DIN; i += 256) {
    int b = i >> 8, d = i & 255;
    sX[i] = (__bf16)x[((size_t)b * NFRM + n) * DIN + d];
  }
  float s1 = s1inv[n], s2 = s2inv[n];
  __syncthreads();

  // ---- H = gelu(X @ W1 * s1 + b1) -----------------------------------------
  {
    v8f acc[2] = {};
    for (int kk = 0; kk < DIN / 32; ++kk) {
      __syncthreads();
      stage_wt_slab<DIN>(w1 + (size_t)n * DIN * DIN + (size_t)kk * 32 * DIN, sWt, t);
      __syncthreads();
      v16bf a = load_frag_a(sX + kk * 32, DIN, lane);
#pragma unroll
      for (int tt = 0; tt < 2; ++tt) {
        v16bf bf = load_frag_bt(sWt + (wv * 2 + tt) * 16 * 32, 32, lane);
        acc[tt] = wmma_bf16(a, bf, acc[tt]);
      }
    }
    int col0 = lane & 15, row0 = (lane >> 4) << 3;
#pragma unroll
    for (int tt = 0; tt < 2; ++tt) {
      int cg = (wv * 2 + tt) * 16 + col0;
      float bias = b1[(size_t)n * DIN + cg];
#pragma unroll
      for (int r = 0; r < 8; ++r) {
        float h = acc[tt][r] * s1 + bias;
        h = 0.5f * h * (1.0f + erff(h * 0.70710678118654752f));  // exact gelu
        sH[(row0 + r) * DIN + cg] = (__bf16)h;
      }
    }
  }

  // ---- Out = H @ W2 * s2 + b2 ---------------------------------------------
  {
    v8f acc[4] = {};
    for (int kk = 0; kk < DIN / 32; ++kk) {
      __syncthreads();
      stage_wt_slab<DOUT>(w2 + (size_t)n * DIN * DOUT + (size_t)kk * 32 * DOUT, sWt, t);
      __syncthreads();
      v16bf a = load_frag_a(sH + kk * 32, DIN, lane);
#pragma unroll
      for (int tt = 0; tt < 4; ++tt) {
        v16bf bf = load_frag_bt(sWt + (wv * 4 + tt) * 16 * 32, 32, lane);
        acc[tt] = wmma_bf16(a, bf, acc[tt]);
      }
    }
    int col0 = lane & 15, row0 = (lane >> 4) << 3;
#pragma unroll
    for (int tt = 0; tt < 4; ++tt) {
      int node = (wv * 4 + tt) * 16 + col0;
      float bias = b2[(size_t)n * DOUT + node];
#pragma unroll
      for (int r = 0; r < 8; ++r) {
        int bb = row0 + r;
        conv[((size_t)bb * NFRM + n) * DOUT + node] = acc[tt][r] * s2 + bias;
      }
    }
  }
}

// ---------------------------------------------------------------------------
// K1b: transpose conv[b][frame][node] -> nf[b][node][frame] (f32 + bf16)
// ---------------------------------------------------------------------------
__global__ __launch_bounds__(256) void kn_nf_transpose(const float* __restrict__ conv,
                                                       float* __restrict__ nf_f32,
                                                       __bf16* __restrict__ nf_bf) {
  int bidx = blockIdx.x;
  int nt = bidx & 15;
  int ft = (bidx >> 4) & 3;
  int b  = bidx >> 6;
  __shared__ float tile[32][33];
  int t = threadIdx.x;
  int tx = t & 31, ty = t >> 5;
  const float* src = conv + ((size_t)b * NFRM + ft * 32) * DOUT + nt * 32;
#pragma unroll
  for (int r = 0; r < 4; ++r) {
    int fr = ty + r * 8;
    tile[fr][tx] = src[(size_t)fr * DOUT + tx];
  }
  __syncthreads();
  size_t obase = ((size_t)b * DOUT + nt * 32) * NFRM + ft * 32;
#pragma unroll
  for (int r = 0; r < 4; ++r) {
    int nd = ty + r * 8;
    float v = tile[tx][nd];
    nf_f32[obase + (size_t)nd * NFRM + tx] = v;
    nf_bf[obase + (size_t)nd * NFRM + tx] = (__bf16)v;
  }
}

// ---------------------------------------------------------------------------
// K2: QK projection. [8192x128] @ [128x2048]; B from transposed wqkT.
// ---------------------------------------------------------------------------
__global__ __launch_bounds__(256) void kn_qkproj(const __bf16* __restrict__ nf,
                                                 const __bf16* __restrict__ wqkT,
                                                 __bf16* __restrict__ qb,
                                                 __bf16* __restrict__ kb) {
  int wg = blockIdx.x * 8 + (threadIdx.x >> 5);
  int lane = threadIdx.x & 31;
  int rt = wg >> 7;
  int ct = wg & 127;
  const __bf16* A  = nf + (size_t)rt * 16 * NFRM;
  const __bf16* Bt = wqkT + (size_t)ct * 16 * NFRM;
  v8f acc = {};
#pragma unroll
  for (int kk = 0; kk < NFRM / 32; ++kk) {
    v16bf a  = load_frag_a(A + kk * 32, NFRM, lane);
    v16bf bf = load_frag_bt(Bt + kk * 32, NFRM, lane);
    acc = wmma_bf16(a, bf, acc);
  }
  int col0 = lane & 15, row0 = (lane >> 4) << 3;
  const float qscale = 0.08838834764831845f;  // 128^-0.5 folded into q
#pragma unroll
  for (int r = 0; r < 8; ++r) {
    int rg = rt * 16 + row0 + r;
    int b = rg >> 9, node = rg & 511;
    int cg = ct * 16 + col0;
    int sel = cg >> 10, ch = (cg >> 7) & 7, f = cg & 127;
    size_t idx = (((size_t)b * NCH + ch) * DOUT + node) * NFRM + f;
    float v = acc[r];
    if (sel == 0) qb[idx] = (__bf16)(v * qscale);
    else          kb[idx] = (__bf16)v;
  }
}

// ---------------------------------------------------------------------------
// K3: attention S = q k^T, row softmax, bf16 attn store, channel-sum in regs.
// ---------------------------------------------------------------------------
__global__ __launch_bounds__(256) void kn_attn(const __bf16* __restrict__ qb,
                                               const __bf16* __restrict__ kb,
                                               __bf16* __restrict__ attn,
                                               float* __restrict__ sumedge) {
  int b  = blockIdx.x >> 5;
  int rt = blockIdx.x & 31;
  int t = threadIdx.x, lane = t & 31, wv = t >> 5;
  __shared__ __bf16 sQ[16 * NFRM];  // 4 KB
  __shared__ float  sS[16 * DOUT];  // 32 KB
  float accSum[2][16];
#pragma unroll
  for (int rr = 0; rr < 2; ++rr)
#pragma unroll
    for (int jj = 0; jj < 16; ++jj) accSum[rr][jj] = 0.0f;

  for (int c = 0; c < NCH; ++c) {
    __syncthreads();
    const __bf16* Q = qb + (((size_t)b * NCH + c) * DOUT + rt * 16) * NFRM;
    ((v8bf*)sQ)[t] = ((const v8bf*)Q)[t];
    __syncthreads();
    const __bf16* K = kb + ((size_t)b * NCH + c) * DOUT * NFRM;
#pragma unroll
    for (int tt = 0; tt < 4; ++tt) {
      int ct = wv + tt * 8;
      v8f acc = {};
#pragma unroll
      for (int kk = 0; kk < NFRM / 32; ++kk) {
        v16bf a  = load_frag_a(sQ + kk * 32, NFRM, lane);
        v16bf bf = load_frag_bt(K + (size_t)(ct * 16) * NFRM + kk * 32, NFRM, lane);
        acc = wmma_bf16(a, bf, acc);
      }
      int col0 = lane & 15, row0 = (lane >> 4) << 3;
#pragma unroll
      for (int r = 0; r < 8; ++r) sS[(row0 + r) * DOUT + ct * 16 + col0] = acc[r];
    }
    __syncthreads();
#pragma unroll
    for (int rr = 0; rr < 2; ++rr) {
      int row = wv * 2 + rr;
      const float* Sr = sS + row * DOUT + lane * 16;
      float vals[16];
#pragma unroll
      for (int q = 0; q < 4; ++q) {
        v4f v = ((const v4f*)Sr)[q];
#pragma unroll
        for (int j = 0; j < 4; ++j) vals[q * 4 + j] = v[j];
      }
      float mx = -1e30f;
#pragma unroll
      for (int jj = 0; jj < 16; ++jj) mx = fmaxf(mx, vals[jj]);
      for (int off = 16; off > 0; off >>= 1) mx = fmaxf(mx, __shfl_xor(mx, off, 32));
      float sum = 0.0f;
#pragma unroll
      for (int jj = 0; jj < 16; ++jj) {
        float e = __expf(vals[jj] - mx);
        vals[jj] = e;
        sum += e;
      }
      for (int off = 16; off > 0; off >>= 1) sum += __shfl_xor(sum, off, 32);
      float inv = 1.0f / sum;
      __bf16* Arow =
          attn + ((((size_t)b * NCH + c) * DOUT) + rt * 16 + row) * DOUT + lane * 16;
#pragma unroll
      for (int h = 0; h < 2; ++h) {
        v8bf o;
#pragma unroll
        for (int j = 0; j < 8; ++j) {
          float a = vals[h * 8 + j] * inv;
          o[j] = (__bf16)a;
          accSum[rr][h * 8 + j] += a;
        }
        ((v8bf*)Arow)[h] = o;
      }
    }
  }
#pragma unroll
  for (int rr = 0; rr < 2; ++rr) {
    int row = wv * 2 + rr;
    float* SE = sumedge + ((size_t)b * DOUT + rt * 16 + row) * DOUT + lane * 16;
#pragma unroll
    for (int q = 0; q < 4; ++q) {
      v4f o = {accSum[rr][q * 4], accSum[rr][q * 4 + 1], accSum[rr][q * 4 + 2],
               accSum[rr][q * 4 + 3]};
      ((v4f*)SE)[q] = o;
    }
  }
}

// ---------------------------------------------------------------------------
// K4: per-(b,row) top-32 threshold of channel-summed attention (wave per row).
// ---------------------------------------------------------------------------
__global__ __launch_bounds__(256) void kn_topk(const float* __restrict__ sumedge,
                                               float* __restrict__ thresh) {
  int wid = blockIdx.x * 8 + (threadIdx.x >> 5);
  int lane = threadIdx.x & 31;
  const float* row = sumedge + (size_t)wid * DOUT + lane * 16;
  float v[16];
#pragma unroll
  for (int q = 0; q < 4; ++q) {
    v4f x = ((const v4f*)row)[q];
#pragma unroll
    for (int j = 0; j < 4; ++j) v[q * 4 + j] = x[j];
  }
  float m = 0.0f;
  for (int it = 0; it < KNEIGH; ++it) {
    float lm = -1e30f;
    int li = 0;
#pragma unroll
    for (int j = 0; j < 16; ++j)
      if (v[j] > lm) { lm = v[j]; li = j; }
    float bm = lm;
    int bl = lane;
    for (int off = 16; off > 0; off >>= 1) {
      float om = __shfl_xor(bm, off, 32);
      int   ol = __shfl_xor(bl, off, 32);
      if (om > bm || (om == bm && ol < bl)) { bm = om; bl = ol; }
    }
    if (lane == bl) {
#pragma unroll
      for (int j = 0; j < 16; ++j)
        if (j == li) v[j] = -1e30f;
    }
    m = bm;
  }
  if (lane == 0) thresh[wid] = m;
}

// ---------------------------------------------------------------------------
// K5: in-place top-k mask (+eye) and row-normalize of attn (wave per row).
// ---------------------------------------------------------------------------
__global__ __launch_bounds__(256) void kn_rownorm(__bf16* __restrict__ attn,
                                                  const float* __restrict__ sumedge,
                                                  const float* __restrict__ thresh) {
  int wid = blockIdx.x * 8 + (threadIdx.x >> 5);
  int lane = threadIdx.x & 31;
  int nrow = wid & 511;
  int bc = wid >> 9;
  int b = bc >> 3;
  __bf16* row = attn + (size_t)wid * DOUT + lane * 16;
  const float* se = sumedge + ((size_t)b * DOUT + nrow) * DOUT + lane * 16;
  float th = thresh[b * DOUT + nrow];
  float sev[16];
#pragma unroll
  for (int q = 0; q < 4; ++q) {
    v4f x = ((const v4f*)se)[q];
#pragma unroll
    for (int j = 0; j < 4; ++j) sev[q * 4 + j] = x[j];
  }
  v8bf a0 = ((const v8bf*)row)[0];
  v8bf a1 = ((const v8bf*)row)[1];
  float vals[16];
  float sum = 0.0f;
#pragma unroll
  for (int j = 0; j < 16; ++j) {
    int col = lane * 16 + j;
    float a = (float)(j < 8 ? a0[j] : a1[j - 8]);
    bool keep = (sev[j] >= th) || (col == nrow);
    a = keep ? a : 0.0f;
    vals[j] = a;
    sum += a;
  }
  for (int off = 16; off > 0; off >>= 1) sum += __shfl_xor(sum, off, 32);
  float inv = 1.0f / (sum + 1e-6f);
#pragma unroll
  for (int h = 0; h < 2; ++h) {
    v8bf o;
#pragma unroll
    for (int j = 0; j < 8; ++j) o[j] = (__bf16)(vals[h * 8 + j] * inv);
    ((v8bf*)row)[h] = o;
  }
}

// ---------------------------------------------------------------------------
// K6: column sums of norm_row per (b,c): 4-way row split + LDS reduce.
// ---------------------------------------------------------------------------
__global__ __launch_bounds__(256) void kn_colsum(const __bf16* __restrict__ attn,
                                                 float* __restrict__ csum) {
  int bc = blockIdx.x;
  int t = threadIdx.x;
  int chunk = t & 63;
  int part = t >> 6;
  const __bf16* M = attn + (size_t)bc * DOUT * DOUT;
  float s[8] = {};
  for (int n = part * 128; n < part * 128 + 128; ++n) {
    v8bf v = *(const v8bf*)(M + (size_t)n * DOUT + chunk * 8);
#pragma unroll
    for (int j = 0; j < 8; ++j) s[j] += (float)v[j];
  }
  __shared__ float red[4][DOUT];
#pragma unroll
  for (int j = 0; j < 8; ++j) red[part][chunk * 8 + j] = s[j];
  __syncthreads();
  if (part == 0) {
#pragma unroll
    for (int j = 0; j < 8; ++j) {
      int c = chunk * 8 + j;
      csum[bc * DOUT + c] = red[0][c] + red[1][c] + red[2][c] + red[3][c];
    }
  }
}

// ---------------------------------------------------------------------------
// K7: edge = (R * inv_colsum) @ R^T per (b,c).  128x128 output blocks, A & B
// K-chunks (128x64 bf16) staged in LDS -> 8x operand reuse vs 16-row tiles.
// Wave w owns one 16-row subtile across 8 col subtiles (A-frag feeds 8 WMMAs).
// ---------------------------------------------------------------------------
__global__ __launch_bounds__(256) void kn_edge(const __bf16* __restrict__ attn,
                                               const float* __restrict__ csum,
                                               float* __restrict__ edge) {
  int id = blockIdx.x;
  int ct = id & 3;         // col block of 128
  int rt = (id >> 2) & 3;  // row block of 128
  int bc = id >> 4;        // (b,c) pair
  int t = threadIdx.x, lane = t & 31, wv = t >> 5;
  __shared__ __bf16 sA[128 * 64];  // 16 KB (scaled A chunk)
  __shared__ __bf16 sB[128 * 64];  // 16 KB
  __shared__ float sInv[DOUT];     // 2 KB
  const __bf16* R = attn + (size_t)bc * DOUT * DOUT;
  for (int i = t; i < DOUT; i += 256) sInv[i] = 1.0f / (csum[bc * DOUT + i] + 1e-6f);
  const int rbase = rt * 128, cbase = ct * 128;
  v8f acc[8] = {};
  __syncthreads();
  for (int kc = 0; kc < 8; ++kc) {  // K chunks of 64
    __syncthreads();
#pragma unroll
    for (int it = 0; it < 4; ++it) {
      int q = it * 256 + t;  // 1024 v8bf chunks per buffer
      int row = q >> 3;
      int k8 = (q & 7) * 8;
      const __bf16* pa = R + (size_t)(rbase + row) * DOUT + kc * 64 + k8;
      const __bf16* pb = R + (size_t)(cbase + row) * DOUT + kc * 64 + k8;
      if (kc < 7) {  // gfx1250 global_prefetch of next K chunk
        __builtin_prefetch(pa + 64, 0, 1);
        __builtin_prefetch(pb + 64, 0, 1);
      }
      v8bf va = *(const v8bf*)pa;
      v8bf vb = *(const v8bf*)pb;
      v8bf oa;
#pragma unroll
      for (int j = 0; j < 8; ++j)
        oa[j] = (__bf16)((float)va[j] * sInv[kc * 64 + k8 + j]);
      *(v8bf*)(sA + row * 64 + k8) = oa;
      *(v8bf*)(sB + row * 64 + k8) = vb;
    }
    __syncthreads();
#pragma unroll
    for (int kh = 0; kh < 2; ++kh) {
      v16bf a = load_frag_a(sA + (wv * 16) * 64 + kh * 32, 64, lane);
#pragma unroll
      for (int ci = 0; ci < 8; ++ci) {
        v16bf bf = load_frag_bt(sB + (ci * 16) * 64 + kh * 32, 64, lane);
        acc[ci] = wmma_bf16(a, bf, acc[ci]);
      }
    }
  }
  int col0 = lane & 15, row0 = (lane >> 4) << 3;
  float* E = edge + (size_t)bc * DOUT * DOUT;
#pragma unroll
  for (int ci = 0; ci < 8; ++ci) {
#pragma unroll
    for (int r = 0; r < 8; ++r)
      E[(size_t)(rbase + wv * 16 + row0 + r) * DOUT + cbase + ci * 16 + col0] =
          acc[ci][r];
  }
}

// ---------------------------------------------------------------------------
extern "C" void kernel_launch(void* const* d_in, const int* in_sizes, int n_in,
                              void* d_out, int out_size, void* d_ws, size_t ws_size,
                              hipStream_t stream) {
  const float* x   = (const float*)d_in[0];  // [16,128,256]
  const float* w1  = (const float*)d_in[1];  // [128,256,256]
  const float* b1  = (const float*)d_in[2];  // [128,1,256]
  const float* w2  = (const float*)d_in[3];  // [128,256,512]
  const float* b2  = (const float*)d_in[4];  // [128,1,512]
  const float* wqk = (const float*)d_in[5];  // [128,2048]

  float* nf_out   = (float*)d_out;                      // [16,512,128]
  float* edge_out = nf_out + (size_t)NB * DOUT * NFRM;  // [16,8,512,512]

  char* wsp = (char*)d_ws;
  auto carve = [&](size_t bytes) -> char* {
    char* p = wsp;
    wsp += (bytes + 255) & ~(size_t)255;
    return p;
  };
  float*  s1inv   = (float*)carve(NFRM * 4);
  float*  s2inv   = (float*)carve(NFRM * 4);
  float*  thr     = (float*)carve((size_t)NB * DOUT * 4);
  float*  csum    = (float*)carve((size_t)NB * NCH * DOUT * 4);
  float*  conv    = (float*)carve((size_t)NB * NFRM * DOUT * 4);
  float*  sumedge = (float*)carve((size_t)NB * DOUT * DOUT * 4);
  __bf16* wqkT    = (__bf16*)carve((size_t)NFRM * 2048 * 2);
  __bf16* nfb     = (__bf16*)carve((size_t)NB * DOUT * NFRM * 2);
  __bf16* qb      = (__bf16*)carve((size_t)NB * NCH * DOUT * NFRM * 2);
  __bf16* kb      = (__bf16*)carve((size_t)NB * NCH * DOUT * NFRM * 2);
  __bf16* attnb   = (__bf16*)carve((size_t)NB * NCH * DOUT * DOUT * 2);

  kn_specnorm<<<256, 256, 0, stream>>>(w1, w2, s1inv, s2inv);
  kn_cvt_wqkT<<<(NFRM * 2048 + 255) / 256, 256, 0, stream>>>(wqk, wqkT);
  kn_mlp<<<NFRM, 256, 0, stream>>>(x, w1, b1, w2, b2, s1inv, s2inv, conv);
  kn_nf_transpose<<<NB * 4 * 16, 256, 0, stream>>>(conv, nf_out, nfb);
  kn_qkproj<<<8192, 256, 0, stream>>>(nfb, wqkT, qb, kb);
  kn_attn<<<NB * 32, 256, 0, stream>>>(qb, kb, attnb, sumedge);
  kn_topk<<<NB * DOUT / 8, 256, 0, stream>>>(sumedge, thr);
  kn_rownorm<<<NB * NCH * DOUT / 8, 256, 0, stream>>>(attnb, sumedge, thr);
  kn_colsum<<<NB * NCH, 256, 0, stream>>>(attnb, csum);
  kn_edge<<<NB * NCH * 4 * 4, 256, 0, stream>>>(attnb, csum, edge_out);
}
